// HopfieldBlock_chemTrainSpace_29420525977875
// MI455X (gfx1250) — matrix operand order repaired and checked by reading, not compiled, not run on hardware
//
#include <hip/hip_runtime.h>
#include <cstdint>
#include <cstddef>

// ---------------------------------------------------------------------------
// Types for CDNA5 WMMA (wave32): bf16 A/B operands (16 x bf16 = 8 VGPRs),
// f32 accumulator (8 x f32 = 8 VGPRs).
// ---------------------------------------------------------------------------
typedef __attribute__((ext_vector_type(16))) __bf16 v16bf;
typedef __attribute__((ext_vector_type(8)))  float  v8f;
typedef __attribute__((ext_vector_type(4)))  int    v4i;

typedef unsigned short u16; // raw bf16 storage for host-visible interfaces

struct alignas(16) F4 { float x, y, z, w; };          // POD 16-byte chunk
struct alignas(16) BF8 { u16 v[8]; };                 // 8 packed bf16

union AFrag { F4 f4[2]; v16bf v; };

// gfx1250 async global->LDS copy path (ASYNCcnt + s_wait_asynccnt)
#if defined(__HIP_DEVICE_COMPILE__) && __has_builtin(__builtin_amdgcn_global_load_async_to_lds_b128)
#define ASYNC_COPY 1
#else
#define ASYNC_COPY 0
#endif

#define ASG __attribute__((address_space(1)))
#define ASL __attribute__((address_space(3)))

static __device__ __forceinline__ void wait_async0() {
#if defined(__HIP_DEVICE_COMPILE__)
#if __has_builtin(__builtin_amdgcn_s_wait_asynccnt)
  __builtin_amdgcn_s_wait_asynccnt(0);
#else
  asm volatile("s_wait_asynccnt 0x0" ::: "memory");
#endif
#endif
}

static __device__ __forceinline__ u16 f2bf(float f) {
  union { float f; uint32_t u; } c; c.f = f;
  uint32_t r = (c.u + 0x7FFFu + ((c.u >> 16) & 1u)) >> 16; // RNE
  return (u16)r;
}

// ---------------------------------------------------------------------------
// Problem constants
// ---------------------------------------------------------------------------
// B=256, T=25 (1 query + 12 actives + 12 inactives), D=1024, H=8, A=512,
// H*A=4096, NR=2048, NROWS = B*T = 6400.
#define NROWS   6400
#define DMODEL  1024
#define HA      4096
#define HEADS   8
#define ADIM    512
#define NRKEYS  2048
#define BETA_F  0.044194173824159216f

// ---------------------------------------------------------------------------
// Elementwise conversion kernels
// ---------------------------------------------------------------------------
__global__ __launch_bounds__(256) void pack_S_bf16(const float* __restrict__ qe,
                                                   const float* __restrict__ ae,
                                                   const float* __restrict__ ie,
                                                   u16* __restrict__ Sf) {
  size_t idx = (size_t)blockIdx.x * 256 + threadIdx.x;   // over NROWS*DMODEL
  int n = (int)(idx >> 10);
  int d = (int)(idx & 1023);
  int b = n / 25, t = n - b * 25;
  float v;
  if (t == 0)       v = qe[(size_t)b * 1024 + d];
  else if (t <= 12) v = ae[(size_t)(b * 12 + (t - 1)) * 1024 + d];
  else              v = ie[(size_t)(b * 12 + (t - 13)) * 1024 + d];
  Sf[idx] = f2bf(v);
}

__global__ __launch_bounds__(256) void cvt_bf16(const float* __restrict__ x,
                                                u16* __restrict__ y) {
  size_t idx = (size_t)blockIdx.x * 256 + threadIdx.x;
  y[idx] = f2bf(x[idx]);
}

// W (Kdim x Ndim, row-major f32) -> WT (Ndim x Kdim, row-major bf16)
__global__ __launch_bounds__(256) void transpose_bf16(const float* __restrict__ W,
                                                      u16* __restrict__ WT,
                                                      int Kdim, int Ndim) {
  size_t idx = (size_t)blockIdx.x * 256 + threadIdx.x;   // over Kdim*Ndim
  int n = (int)(idx % Ndim);
  int k = (int)(idx / Ndim);
  WT[(size_t)n * Kdim + k] = f2bf(W[idx]);
}

// ---------------------------------------------------------------------------
// Tiled WMMA GEMM:  C(M,N) = alpha * A(M,K) * Bt(N,K)^T  (+ bias[n])
// Block tile 128x128, 256 threads (8 waves), each wave computes 32x64.
// Double-buffered LDS; next K-slab staged with GLOBAL_LOAD_ASYNC_TO_LDS_B128
// (ASYNCcnt) while the current slab feeds the WMMA pipe.
// EP = 0: store bf16, row-major ldc
// EP = 1: store bf16 transposed (C^T, ldc = M-stride)
// EP = 2: store f32, row-major ldc
// EP = 3: add residual (gathered from 3 input tensors) and scatter-store f32
//         into the 3 concatenated output chunks.
// ---------------------------------------------------------------------------
template<int EP>
__global__ __launch_bounds__(256) void gemm_wmma(
    const u16* __restrict__ A, int lda,
    const u16* __restrict__ Bt, int ldbt,
    const float* __restrict__ bias, float alpha, int K,
    void* __restrict__ Cv, int ldc,
    const float* __restrict__ r0, const float* __restrict__ r1,
    const float* __restrict__ r2) {
  // LDS tiles, padded row stride of 40 elements (80 B) to break bank conflicts
  __shared__ u16 ldsA[2][128 * 40];
  __shared__ u16 ldsB[2][128 * 40];

  const int tid  = threadIdx.x;
  const int lane = tid & 31;
  const int wave = tid >> 5;
  const int mb = (wave >> 1) * 32;   // wave's row offset in block tile
  const int nb = (wave & 1) * 64;    // wave's col offset in block tile
  const int mr = lane & 15;          // row/col within 16x16 tile
  const int kh = lane >> 4;          // which K-half this lane holds
  const int blockM = blockIdx.y * 128;
  const int blockN = blockIdx.x * 128;

  v8f acc[2][4];
  for (int i = 0; i < 2; i++)
    for (int j = 0; j < 4; j++)
      acc[i][j] = (v8f){0.f, 0.f, 0.f, 0.f, 0.f, 0.f, 0.f, 0.f};

  // Compute one 32-deep K-slab out of LDS buffer `buf`.
  auto do_tile = [&](int buf) {
    AFrag a[2], b[4];
    for (int rt = 0; rt < 2; rt++) {
      int r = mb + rt * 16 + mr;
      a[rt].f4[0] = *(F4*)&ldsA[buf][r * 40 + kh * 8];        // K = kh*8..+7
      a[rt].f4[1] = *(F4*)&ldsA[buf][r * 40 + 16 + kh * 8];   // K = 16+kh*8..
    }
    for (int ct = 0; ct < 4; ct++) {
      int n = nb + ct * 16 + mr;
      b[ct].f4[0] = *(F4*)&ldsB[buf][n * 40 + kh * 8];        // K = kh*8..+7
      b[ct].f4[1] = *(F4*)&ldsB[buf][n * 40 + 16 + kh * 8];   // K = 16+kh*8..
    }
    for (int rt = 0; rt < 2; rt++)
      for (int ct = 0; ct < 4; ct++)
        acc[rt][ct] = __builtin_amdgcn_wmma_f32_16x16x32_bf16(
            false, a[rt].v, false, b[ct].v, (short)0, acc[rt][ct], false, false);
  };

#if ASYNC_COPY
  // Issue this thread's share (4 x 16B) of the A/B tile as async LDS copies.
  // Builtin signature (from compiler diagnostics):
  //   (v4i addrspace(1)* src, v4i addrspace(3)* dst, imm offset, imm cpol)
  auto stage_async = [&](int buf, int kb) {
    for (int i = 0; i < 2; i++) {
      int c = tid + i * 256;                 // 512 16B-chunks per operand
      int row = c >> 2, col8 = (c & 3) * 8;
      __builtin_amdgcn_global_load_async_to_lds_b128(
          (ASG v4i*)(A + (size_t)(blockM + row) * lda + kb + col8),
          (ASL v4i*)&ldsA[buf][row * 40 + col8], 0, 0);
      __builtin_amdgcn_global_load_async_to_lds_b128(
          (ASG v4i*)(Bt + (size_t)(blockN + row) * ldbt + kb + col8),
          (ASL v4i*)&ldsB[buf][row * 40 + col8], 0, 0);
    }
  };

  stage_async(0, 0);
  wait_async0();
  __syncthreads();
  int buf = 0;
  for (int kb = 0; kb < K; kb += 32) {
    if (kb + 32 < K) stage_async(buf ^ 1, kb + 32);  // overlap with WMMAs
    do_tile(buf);
    wait_async0();
    __syncthreads();
    buf ^= 1;
  }
#else
  // Fallback: register-batched staging (all loads issued, then all stores).
  auto load_regs = [&](int kb, F4* ra, F4* rb) {
    for (int i = 0; i < 2; i++) {
      int c = tid + i * 256;
      int row = c >> 2, col8 = (c & 3) * 8;
      ra[i] = *(const F4*)&A[(size_t)(blockM + row) * lda + kb + col8];
      rb[i] = *(const F4*)&Bt[(size_t)(blockN + row) * ldbt + kb + col8];
    }
  };
  auto store_lds = [&](int buf, const F4* ra, const F4* rb) {
    for (int i = 0; i < 2; i++) {
      int c = tid + i * 256;
      int row = c >> 2, col8 = (c & 3) * 8;
      *(F4*)&ldsA[buf][row * 40 + col8] = ra[i];
      *(F4*)&ldsB[buf][row * 40 + col8] = rb[i];
    }
  };

  {
    F4 ra[2], rb[2];
    load_regs(0, ra, rb);
    store_lds(0, ra, rb);
  }
  __syncthreads();
  int buf = 0;
  for (int kb = 0; kb < K; kb += 32) {
    F4 ra[2], rb[2];
    bool pending = (kb + 32 < K);
    if (pending) {
      load_regs(kb + 32, ra, rb);          // global loads overlap with WMMAs
      int row = tid >> 2, col8 = (tid & 3) * 8;
      __builtin_prefetch(&A[(size_t)(blockM + row) * lda + kb + 64 + col8], 0, 3);
      __builtin_prefetch(&Bt[(size_t)(blockN + row) * ldbt + kb + 64 + col8], 0, 3);
    }
    do_tile(buf);
    if (pending) store_lds(buf ^ 1, ra, rb);
    __syncthreads();
    buf ^= 1;
  }
#endif

  // Epilogue. C/D layout: VGPR r holds row (kh*8 + r) of the 16x16 tile,
  // column = lane & 15.
  for (int rt = 0; rt < 2; rt++) {
    for (int ct = 0; ct < 4; ct++) {
      int mrow0 = blockM + mb + rt * 16 + kh * 8;
      int ncol  = blockN + nb + ct * 16 + mr;
      float bn = bias ? bias[ncol] : 0.0f;
      v8f c = acc[rt][ct];
      if (EP == 1) {
        BF8 pk;
        for (int r = 0; r < 8; r++) pk.v[r] = f2bf(c[r] * alpha + bn);
        *(BF8*)((u16*)Cv + (size_t)ncol * ldc + mrow0) = pk;  // one 16B store
      } else {
        for (int r = 0; r < 8; r++) {
          int gm = mrow0 + r;
          float v = c[r] * alpha + bn;
          if (EP == 0) {
            ((u16*)Cv)[(size_t)gm * ldc + ncol] = f2bf(v);
          } else if (EP == 2) {
            ((float*)Cv)[(size_t)gm * ldc + ncol] = v;
          } else { // EP == 3: residual + scatter into 3 output chunks
            int bidx = gm / 25, t = gm - bidx * 25;
            float res; float* dst;
            if (t == 0) {
              size_t o = (size_t)bidx * 1024 + ncol;
              res = r0[o]; dst = (float*)Cv + o;
            } else if (t <= 12) {
              size_t o = (size_t)(bidx * 12 + (t - 1)) * 1024 + ncol;
              res = r1[o]; dst = (float*)Cv + 262144 + o;
            } else {
              size_t o = (size_t)(bidx * 12 + (t - 13)) * 1024 + ncol;
              res = r2[o]; dst = (float*)Cv + 262144 + 3145728 + o;
            }
            *dst = v + res;
          }
        }
      }
    }
  }
}

// ---------------------------------------------------------------------------
// Row softmax over 2048 columns; output bf16 probabilities.
// One 256-thread block (8 wave32s) per row.
// ---------------------------------------------------------------------------
__global__ __launch_bounds__(256) void softmax_rows(const float* __restrict__ S,
                                                    u16* __restrict__ P) {
  const int row = blockIdx.x;
  const int tid = threadIdx.x;
  const float* s = S + (size_t)row * NRKEYS;
  __shared__ float redm[8];
  __shared__ float reds[8];

  float x[8];
  float m = -3.4e38f;
  for (int i = 0; i < 8; i++) {
    x[i] = s[tid + i * 256];
    m = fmaxf(m, x[i]);
  }
  for (int off = 16; off > 0; off >>= 1) m = fmaxf(m, __shfl_xor(m, off, 32));
  if ((tid & 31) == 0) redm[tid >> 5] = m;
  __syncthreads();
  m = redm[0];
  for (int w = 1; w < 8; w++) m = fmaxf(m, redm[w]);

  float sum = 0.f;
  for (int i = 0; i < 8; i++) { x[i] = __expf(x[i] - m); sum += x[i]; }
  for (int off = 16; off > 0; off >>= 1) sum += __shfl_xor(sum, off, 32);
  if ((tid & 31) == 0) reds[tid >> 5] = sum;
  __syncthreads();
  sum = 0.f;
  for (int w = 0; w < 8; w++) sum += reds[w];
  float inv = 1.0f / sum;

  u16* p = P + (size_t)row * NRKEYS;
  for (int i = 0; i < 8; i++) p[tid + i * 256] = f2bf(x[i] * inv);
}

// ---------------------------------------------------------------------------
// Host driver
// ---------------------------------------------------------------------------
extern "C" void kernel_launch(void* const* d_in, const int* in_sizes, int n_in,
                              void* d_out, int out_size, void* d_ws, size_t ws_size,
                              hipStream_t stream) {
  (void)in_sizes; (void)n_in; (void)out_size; (void)ws_size;
  const float* qe  = (const float*)d_in[0];
  const float* ae  = (const float*)d_in[1];
  const float* ie  = (const float*)d_in[2];
  const float* ref = (const float*)d_in[5];
  const float* Wq  = (const float*)d_in[6];
  const float* bq  = (const float*)d_in[7];
  const float* Wk  = (const float*)d_in[8];
  const float* bk  = (const float*)d_in[9];
  const float* Wv  = (const float*)d_in[10];
  const float* bv  = (const float*)d_in[11];
  const float* Wo  = (const float*)d_in[12];
  const float* bo  = (const float*)d_in[13];

  char* wsb = (char*)d_ws;
  size_t off = 0;
  auto alloc = [&](size_t bytes) -> void* {
    void* p = wsb + off;
    off += (bytes + 255) & ~(size_t)255;
    return p;
  };
  u16*   Sf   = (u16*)alloc((size_t)NROWS * DMODEL * 2);   // S gathered, bf16
  u16*   refb = (u16*)alloc((size_t)NRKEYS * DMODEL * 2);  // ref bf16
  u16*   WqT  = (u16*)alloc((size_t)HA * DMODEL * 2);      // Wq^T (N x K)
  u16*   WkT  = (u16*)alloc((size_t)HA * DMODEL * 2);
  u16*   WvT  = (u16*)alloc((size_t)HA * DMODEL * 2);
  u16*   WoT  = (u16*)alloc((size_t)DMODEL * HA * 2);      // Wo^T (1024 x 4096)
  u16*   qb   = (u16*)alloc((size_t)NROWS * HA * 2);       // q bf16 (6400x4096)
  u16*   kbf  = (u16*)alloc((size_t)NRKEYS * HA * 2);      // k bf16 (2048x4096)
  u16*   vT   = (u16*)alloc((size_t)HA * NRKEYS * 2);      // v^T bf16 (4096x2048)
  float* sc   = (float*)alloc((size_t)NROWS * NRKEYS * 4); // per-head scores f32
  u16*   P    = (u16*)alloc((size_t)NROWS * NRKEYS * 2);   // per-head probs bf16
  u16*   ao   = (u16*)alloc((size_t)NROWS * HA * 2);       // attn output bf16

  // --- conversions ---
  pack_S_bf16<<<(NROWS * DMODEL) / 256, 256, 0, stream>>>(qe, ae, ie, Sf);
  cvt_bf16<<<(NRKEYS * DMODEL) / 256, 256, 0, stream>>>(ref, refb);
  transpose_bf16<<<(DMODEL * HA) / 256, 256, 0, stream>>>(Wq, WqT, DMODEL, HA);
  transpose_bf16<<<(DMODEL * HA) / 256, 256, 0, stream>>>(Wk, WkT, DMODEL, HA);
  transpose_bf16<<<(DMODEL * HA) / 256, 256, 0, stream>>>(Wv, WvT, DMODEL, HA);
  transpose_bf16<<<(HA * DMODEL) / 256, 256, 0, stream>>>(Wo, WoT, HA, DMODEL);

  // --- projections (bf16 in, f32 accumulate, bf16 out) ---
  // q = S @ Wq + bq : M=6400 N=4096 K=1024
  gemm_wmma<0><<<dim3(HA / 128, NROWS / 128), 256, 0, stream>>>(
      Sf, DMODEL, WqT, DMODEL, bq, 1.0f, DMODEL, qb, HA, nullptr, nullptr, nullptr);
  // k = ref @ Wk + bk : M=2048 N=4096 K=1024
  gemm_wmma<0><<<dim3(HA / 128, NRKEYS / 128), 256, 0, stream>>>(
      refb, DMODEL, WkT, DMODEL, bk, 1.0f, DMODEL, kbf, HA, nullptr, nullptr, nullptr);
  // v^T = (ref @ Wv + bv)^T : stored transposed so PV GEMM's B is N x K
  gemm_wmma<1><<<dim3(HA / 128, NRKEYS / 128), 256, 0, stream>>>(
      refb, DMODEL, WvT, DMODEL, bv, 1.0f, DMODEL, vT, NRKEYS, nullptr, nullptr, nullptr);

  // --- per-head attention (scores + probs fit in 192MB L2) ---
  for (int h = 0; h < HEADS; h++) {
    // scores = BETA * q_h @ k_h^T : M=6400 N=2048 K=512
    gemm_wmma<2><<<dim3(NRKEYS / 128, NROWS / 128), 256, 0, stream>>>(
        qb + h * ADIM, HA, kbf + h * ADIM, HA, nullptr, BETA_F, ADIM,
        sc, NRKEYS, nullptr, nullptr, nullptr);
    softmax_rows<<<NROWS, 256, 0, stream>>>(sc, P);
    // out_h = P @ v_h : M=6400 N=512 K=2048
    gemm_wmma<0><<<dim3(ADIM / 128, NROWS / 128), 256, 0, stream>>>(
        P, NRKEYS, vT + (size_t)h * ADIM * NRKEYS, NRKEYS, nullptr, 1.0f, NRKEYS,
        ao + h * ADIM, HA, nullptr, nullptr, nullptr);
  }

  // --- output projection + residual + scatter to the 3 output chunks ---
  // S_h = attn_out @ Wo + bo ; out = S + S_h : M=6400 N=1024 K=4096
  gemm_wmma<3><<<dim3(DMODEL / 128, NROWS / 128), 256, 0, stream>>>(
      ao, HA, WoT, HA, bo, 1.0f, HA, d_out, 0, qe, ae, ie);
}
